// MultiHeadAttentionBlock_87866440941863
// MI455X (gfx1250) — matrix operand order repaired and checked by reading, not compiled, not run on hardware
//
#include <hip/hip_runtime.h>

typedef __attribute__((ext_vector_type(16))) __bf16 v16bf;
typedef __attribute__((ext_vector_type(8)))  __bf16 v8bf;
typedef __attribute__((ext_vector_type(8)))  float  v8f;
typedef unsigned int u32x4 __attribute__((ext_vector_type(4)));
typedef int          i32x8 __attribute__((ext_vector_type(8)));
typedef int          i32x4 __attribute__((ext_vector_type(4)));

#define B_  2
#define L_  2048
#define DM_ 1024
#define H_  16
#define DK_ 64

#define KC_ 32     // K step per iteration (one WMMA depth)
#define BM_ 128    // block tile M: 8 waves x 16 rows
#define BN_ 64     // block tile N: 4 WMMA tiles per wave

__device__ __forceinline__ __bf16 to_bf16(float f) {
  union { float f; unsigned u; } uf; uf.f = f;
  unsigned u = uf.u;
  u += 0x7FFFu + ((u >> 16) & 1u);   // round-to-nearest-even
  unsigned short hs = (unsigned short)(u >> 16);
  __bf16 r;
  __builtin_memcpy(&r, &hs, 2);
  return r;
}

// ---------------------------------------------------------------------------
// fp32 -> bf16 cast
// ---------------------------------------------------------------------------
__global__ __launch_bounds__(256)
void cast_bf16(const float* __restrict__ in, __bf16* __restrict__ out, int n) {
  int i = blockIdx.x * blockDim.x + threadIdx.x;
  if (i < n) out[i] = to_bf16(in[i]);
}

// ---------------------------------------------------------------------------
// WMMA GEMM:  C[M,N] = scale * A[M,K] * W[N,K]^T     (row strides == K)
//
// Block = 256 threads = 8 waves, computes a BM_ x BN_ (128x64) tile.
// Wave w computes rows [m0+16w, +16) x all 64 columns: 4 accumulators,
// A fragment loaded once per k-step and reused across 4 WMMAs.
// The shared 64x32 bf16 B panel is staged in LDS by the Tensor Data Mover,
// double-buffered: the DMA for panel i+1 is issued right after barrier i and
// overlaps the WMMA work on panel i (one tensor op in flight, one barrier
// per k-step).
//
// MODE 0: store bf16, split heads  [B,H,L,DK]   (Q/K projection)
// MODE 1: store bf16, V transposed [B,H,DK,L]   (V projection)
// MODE 2: store bf16, row-major with ldc        (attention out)
// MODE 3: store f32,  row-major with ldc        (scores / final proj)
// ---------------------------------------------------------------------------
template <int MODE>
__global__ __launch_bounds__(256)
void wmma_gemm(const __bf16* __restrict__ A, const __bf16* __restrict__ W,
               void* __restrict__ C, int M, int N, int K, int ldc, float scale)
{
  __shared__ __attribute__((aligned(64))) __bf16 Bsh[2][BN_ * KC_];  // 2 x 4 KB

  const int lane  = threadIdx.x & 31;
  const int wavei = threadIdx.x >> 5;
  const int nBlk  = N / BN_;
  const int bm    = blockIdx.x / nBlk;
  const int bn    = blockIdx.x - bm * nBlk;
  const int m0    = bm * BM_ + wavei * 16;
  const int n0    = bn * BN_;

  // A fragment (16x32 bf16): lane -> row (lane&15);
  // lanes 0-15 hold K [0..15], lanes 16-31 hold K [16..31].
  const int arow = m0 + (lane & 15);
  const int kbA  = (lane >> 4) << 4;
  const __bf16* __restrict__ Ap = A + (size_t)arow * K + kbA;

  // B fragment (32x16 bf16) from LDS: lane -> column (lane&15);
  // lanes 0-15 hold K [0..15], lanes 16-31 hold K [16..31].
  const int brow = lane & 15;
  const int kbB  = (lane >> 4) << 4;

  v8f c[4] = {{}, {}, {}, {}};

#if __has_builtin(__builtin_amdgcn_tensor_load_to_lds)
  const unsigned lds0 = (unsigned)(size_t)(&Bsh[0][0]);   // LDS byte offsets
  const unsigned lds1 = (unsigned)(size_t)(&Bsh[1][0]);
  const unsigned long long gbase =
      (unsigned long long)(size_t)W + (unsigned long long)n0 * K * 2ull;

  // Tensor DMA descriptor (CDNA5 ISA): 2D tile,
  // tile_dim0 = KC (contiguous K), tile_dim1 = BN rows, stride = K.
  auto tdm_issue = [&](int kk, unsigned lds_off) {
    const unsigned long long ga = gbase + (unsigned long long)kk * 2ull;
    u32x4 g0;
    g0[0] = 1u;                                            // count=1 (valid)
    g0[1] = lds_off;                                       // lds_addr
    g0[2] = (unsigned)(ga & 0xFFFFFFFFull);                // global_addr lo
    g0[3] = (unsigned)((ga >> 32) & 0x01FFFFFFull)         // global_addr hi
            | (2u << 30);                                  // type=2 (image)
    i32x8 g1;
    g1[0] = (int)(1u << 16);                               // data_size=2B
    g1[1] = (int)(((unsigned)K & 0xFFFFu) << 16);          // tensor_dim0 lo16
    g1[2] = (int)((((unsigned)K >> 16) & 0xFFFFu)
            | (((unsigned)N & 0xFFFFu) << 16));            // dim0 hi | dim1 lo
    g1[3] = (int)((((unsigned)N >> 16) & 0xFFFFu)
            | ((unsigned)KC_ << 16));                      // dim1 hi | tile_dim0
    g1[4] = (int)(unsigned)BN_;                            // tile_dim1; tile_dim2=0
    g1[5] = (int)(unsigned)K;                              // dim0_stride lo32
    g1[6] = 0;                                             // stride hi | dim1_stride lo
    g1[7] = 0;
    i32x4 gz4 = {0, 0, 0, 0};                              // 2D: groups 2/3 unused
    i32x8 gz8 = {0, 0, 0, 0, 0, 0, 0, 0};
    __builtin_amdgcn_tensor_load_to_lds(g0, g1, gz4, gz4, gz8, 0);
  };

  if (wavei == 0) tdm_issue(0, lds0);
#endif

  int ib = 0;
  for (int k0 = 0; k0 < K; k0 += KC_, ib ^= 1) {
#if __has_builtin(__builtin_amdgcn_tensor_load_to_lds)
    if (wavei == 0) __builtin_amdgcn_s_wait_tensorcnt(0);  // panel ib ready
    __syncthreads();                                       // visible to all;
                                                           // panel ib^1 free
    if (wavei == 0 && (k0 + KC_ < K))
      tdm_issue(k0 + KC_, ib ? lds0 : lds1);               // overlap with WMMAs
#else
    {   // fallback: cooperative global->LDS copy (one b128 chunk per thread)
      const int chunk = threadIdx.x;          // 0..255 == 64 rows x 4 chunks
      const int row   = chunk >> 2;
      const int cpos  = (chunk & 3) * 8;
      *(v8bf*)(&Bsh[ib][row * KC_ + cpos]) =
          *(const v8bf*)(W + (size_t)(n0 + row) * K + k0 + cpos);
    }
    __syncthreads();
#endif

    v8bf alo = *(const v8bf*)(Ap + k0);
    v8bf ahi = *(const v8bf*)(Ap + k0 + 8);
    v16bf a;
#pragma unroll
    for (int i = 0; i < 8; ++i) { a[i] = alo[i]; a[i + 8] = ahi[i]; }
    if (k0 + KC_ < K) __builtin_prefetch(Ap + k0 + KC_, 0, 3);

#pragma unroll
    for (int j = 0; j < 4; ++j) {
      v16bf b = *(const v16bf*)(&Bsh[ib][(size_t)(j * 16 + brow) * KC_ + kbB]);
      c[j] = __builtin_amdgcn_wmma_f32_16x16x32_bf16(
          false, a, false, b, (short)0, c[j], false, false);
    }

#if !__has_builtin(__builtin_amdgcn_tensor_load_to_lds)
    __syncthreads();
#endif
  }

  // C layout: lane&15 -> column, rows (lane>>4)*8 + r in VGPR r.
  const int cn = lane & 15;
  const int mb = (lane >> 4) << 3;
#pragma unroll
  for (int j = 0; j < 4; ++j) {
#pragma unroll
    for (int r = 0; r < 8; ++r) {
      const int gm = m0 + mb + r;
      const int gn = n0 + j * 16 + cn;
      const float val = c[j][r] * scale;
      if (MODE == 3) {
        ((float*)C)[(size_t)gm * ldc + gn] = val;
      } else {
        const __bf16 bv = to_bf16(val);
        if (MODE == 2) {
          ((__bf16*)C)[(size_t)gm * ldc + gn] = bv;
        } else {
          const int bb = gm >> 11, l = gm & (L_ - 1);   // gm = b*L + l
          const int h  = gn >> 6,  d = gn & (DK_ - 1);  // gn = h*DK + d
          if (MODE == 0)
            ((__bf16*)C)[(((size_t)(bb * H_ + h) * L_) + l) * DK_ + d] = bv;
          else  // MODE 1: V transposed [B,H,DK,L]
            ((__bf16*)C)[(((size_t)(bb * H_ + h) * DK_) + d) * L_ + l] = bv;
        }
      }
    }
  }
}

// ---------------------------------------------------------------------------
// Exact 1.5-entmax over a row of length L via bisection on
//   f(tau) = sum_i [(z_i - tau)_+]^2 = 1,  monotone decreasing, tau in [-1,0].
// One 256-thread block per row; 8 elements per thread held in registers.
// ---------------------------------------------------------------------------
__global__ __launch_bounds__(256)
void entmax15_kernel(const float* __restrict__ S, const int* __restrict__ mask,
                     __bf16* __restrict__ P)
{
  const int q = blockIdx.x;
  const int t = threadIdx.x;
  const float* srow = S + (size_t)q * L_;
  const int*   mrow = mask + (size_t)q * L_;

  float z[8];
  float mx = -3.402823466e38f;
#pragma unroll
  for (int i = 0; i < 8; ++i) {
    const int idx = t + i * 256;
    float s = srow[idx];
    if (mrow[idx] == 0) s = -3.402823466e38f;
    z[i] = s;
    mx = fmaxf(mx, s);
  }

  __shared__ float red[256];
  red[t] = mx;
  __syncthreads();
  for (int off2 = 128; off2 > 0; off2 >>= 1) {
    if (t < off2) red[t] = fmaxf(red[t], red[t + off2]);
    __syncthreads();
  }
  mx = red[0];
  __syncthreads();

#pragma unroll
  for (int i = 0; i < 8; ++i) z[i] = (z[i] - mx) * 0.5f;

  float lo = -1.0f, hi = 0.0f;
  for (int it = 0; it < 30; ++it) {
    const float tau = 0.5f * (lo + hi);
    float s = 0.0f;
#pragma unroll
    for (int i = 0; i < 8; ++i) {
      float d = fmaxf(z[i] - tau, 0.0f);
      s += d * d;
    }
    red[t] = s;
    __syncthreads();
    for (int off2 = 128; off2 > 0; off2 >>= 1) {
      if (t < off2) red[t] += red[t + off2];
      __syncthreads();
    }
    s = red[0];
    __syncthreads();
    if (s >= 1.0f) lo = tau; else hi = tau;
  }

  const float tau = 0.5f * (lo + hi);
#pragma unroll
  for (int i = 0; i < 8; ++i) {
    float d = fmaxf(z[i] - tau, 0.0f);
    P[(size_t)q * L_ + t + i * 256] = to_bf16(d * d);
  }
}

// ---------------------------------------------------------------------------
// Host driver
// ---------------------------------------------------------------------------
extern "C" void kernel_launch(void* const* d_in, const int* in_sizes, int n_in,
                              void* d_out, int out_size, void* d_ws, size_t ws_size,
                              hipStream_t stream)
{
  const float* q_in  = (const float*)d_in[0];
  const float* k_in  = (const float*)d_in[1];
  const float* v_in  = (const float*)d_in[2];
  const int*   mask  = (const int*)d_in[3];
  const float* wq    = (const float*)d_in[4];
  const float* wk    = (const float*)d_in[5];
  const float* wv    = (const float*)d_in[6];
  const float* wo    = (const float*)d_in[7];

  const size_t nBL = (size_t)B_ * L_ * DM_;        // 4 194 304
  const size_t nW  = (size_t)DM_ * DM_;            // 1 048 576
  const size_t nH  = (size_t)B_ * H_ * L_ * DK_;   // 4 194 304
  const size_t nP  = (size_t)L_ * L_;              // 4 194 304

  char* ws = (char*)d_ws;
  size_t off = 0;
  auto take = [&](size_t bytes) -> char* {
    char* p = ws + off;
    off += (bytes + 255) & ~(size_t)255;
    return p;
  };
  __bf16* qb   = (__bf16*)take(nBL * 2);
  __bf16* kb   = (__bf16*)take(nBL * 2);
  __bf16* vb   = (__bf16*)take(nBL * 2);
  __bf16* wqb  = (__bf16*)take(nW * 2);
  __bf16* wkb  = (__bf16*)take(nW * 2);
  __bf16* wvb  = (__bf16*)take(nW * 2);
  __bf16* wob  = (__bf16*)take(nW * 2);
  __bf16* Qh   = (__bf16*)take(nH * 2);   // [B,H,L,DK]
  __bf16* Kh   = (__bf16*)take(nH * 2);   // [B,H,L,DK]
  __bf16* Vt   = (__bf16*)take(nH * 2);   // [B,H,DK,L]
  __bf16* attn = (__bf16*)take(nBL * 2);  // [B*L, DM]
  __bf16* prob = (__bf16*)take(nP * 2);   // [L, L] slice (reused)
  float*  S    = (float*)take(nP * 4);    // [L, L] slice (reused)

  // 1) fp32 -> bf16 casts
  cast_bf16<<<(int)(nBL / 256), 256, 0, stream>>>(q_in, qb, (int)nBL);
  cast_bf16<<<(int)(nBL / 256), 256, 0, stream>>>(k_in, kb, (int)nBL);
  cast_bf16<<<(int)(nBL / 256), 256, 0, stream>>>(v_in, vb, (int)nBL);
  cast_bf16<<<(int)(nW  / 256), 256, 0, stream>>>(wq, wqb, (int)nW);
  cast_bf16<<<(int)(nW  / 256), 256, 0, stream>>>(wk, wkb, (int)nW);
  cast_bf16<<<(int)(nW  / 256), 256, 0, stream>>>(wv, wvb, (int)nW);
  cast_bf16<<<(int)(nW  / 256), 256, 0, stream>>>(wo, wob, (int)nW);

  // 2) Q/K/V projections: [4096,1024] = [4096,1024] x [1024,1024]^T
  {
    const int M = B_ * L_, N = DM_, K = DM_;
    const int blocks = (M / BM_) * (N / BN_);
    wmma_gemm<0><<<blocks, 256, 0, stream>>>(qb, wqb, Qh, M, N, K, 0, 1.0f);
    wmma_gemm<0><<<blocks, 256, 0, stream>>>(kb, wkb, Kh, M, N, K, 0, 1.0f);
    wmma_gemm<1><<<blocks, 256, 0, stream>>>(vb, wvb, Vt, M, N, K, 0, 1.0f);
  }

  // 3) Per-(b,h) attention: scores -> entmax1.5 -> P*V  (slice buffers reused;
  //    same-stream ordering serializes the reuse correctly)
  for (int b = 0; b < B_; ++b) {
    for (int h = 0; h < H_; ++h) {
      const __bf16* Qs = Qh + (size_t)(b * H_ + h) * L_ * DK_;
      const __bf16* Ks = Kh + (size_t)(b * H_ + h) * L_ * DK_;
      const __bf16* Vs = Vt + (size_t)(b * H_ + h) * DK_ * L_;

      {   // scores = (Q Kt) / sqrt(dk) : [2048,2048] fp32
        const int M = L_, N = L_, K = DK_;
        const int blocks = (M / BM_) * (N / BN_);
        wmma_gemm<3><<<blocks, 256, 0, stream>>>(Qs, Ks, S, M, N, K, L_, 0.125f);
      }

      entmax15_kernel<<<L_, 256, 0, stream>>>(S, mask + (size_t)b * L_ * L_, prob);

      {   // attn[:, h*DK:(h+1)*DK] = P * V  via W = Vt
        const int M = L_, N = DK_, K = L_;
        const int blocks = (M / BM_) * (N / BN_);
        __bf16* Co = attn + (size_t)b * L_ * DM_ + h * DK_;
        wmma_gemm<2><<<blocks, 256, 0, stream>>>(prob, Vs, Co, M, N, K, DM_, 1.0f);
      }
    }
  }

  // 4) Final projection -> fp32 output
  {
    const int M = B_ * L_, N = DM_, K = DM_;
    const int blocks = (M / BM_) * (N / BN_);
    wmma_gemm<3><<<blocks, 256, 0, stream>>>(attn, wob, d_out, M, N, K, DM_, 1.0f);
  }

  (void)in_sizes; (void)n_in; (void)out_size; (void)ws_size;
}
